// Block_19524921327813
// MI455X (gfx1250) — compile-verified, matrix operand order checked by reading, not run on hardware
//
#include <hip/hip_runtime.h>
#include <stdint.h>

typedef __bf16 bf16;
typedef unsigned short u16;
typedef __attribute__((ext_vector_type(8)))  __bf16 bf16x8;
typedef __attribute__((ext_vector_type(16))) __bf16 bf16x16;
typedef __attribute__((ext_vector_type(8)))  float  f32x8;

#define DEV_INLINE __device__ __forceinline__

constexpr int kB = 4, kS = 2048, kD = 1024, kH = 16, kHD = 64, kCAP = 1024;
constexpr float kLnEps = 1e-5f;
constexpr float kLog10000 = 9.210340371976184f;

#define WMMA_BF16(a, b, c) \
  __builtin_amdgcn_wmma_f32_16x16x32_bf16(false, (a), false, (b), (short)0, (c), false, false)

DEV_INLINE f32x8 zero8() {
  f32x8 v;
#pragma unroll
  for (int i = 0; i < 8; ++i) v[i] = 0.0f;
  return v;
}

// LDS byte address of a __shared__ element: low 32 bits of the generic pointer
// (gfx1250 maps generic->LDS by discarding the upper bits).
DEV_INLINE uint32_t lds32(const void* p) { return (uint32_t)(uintptr_t)p; }

// Async global->LDS copy, 16 bytes per lane; tracked with ASYNCcnt.
DEV_INLINE void async_ld_b128(uint32_t lds_off, const void* gaddr) {
  asm volatile("global_load_async_to_lds_b128 %0, %1, off"
               :
               : "v"(lds_off), "v"(gaddr)
               : "memory");
}

// Build a 16x32 bf16 A/B fragment half-pair from LDS (row base, per-lane K offset).
// Lanes 0-15: K {0..7} then {16..23}; lanes 16-31: K {8..15} then {24..31}.
DEV_INLINE bf16x16 make_frag(const bf16* base, int koff) {
  bf16x8 lo = *(const bf16x8*)(base + koff);
  bf16x8 hi = *(const bf16x8*)(base + 16 + koff);
  return __builtin_shufflevector(lo, hi, 0, 1, 2, 3, 4, 5, 6, 7, 8, 9, 10, 11, 12, 13, 14, 15);
}

// ---------------------------------------------------------------------------
// Router: logits -> softmax over S -> bitonic top-k (CAP of S) per batch
// ---------------------------------------------------------------------------
__global__ __launch_bounds__(1024) void router_topk_kernel(
    const float* __restrict__ seq, const float* __restrict__ rw_g,
    float* __restrict__ topw, int* __restrict__ topi) {
  __shared__ float vals[kS];
  __shared__ int   idxs[kS];
  __shared__ float rw[kD];
  __shared__ float red[32];
  const int b = blockIdx.x, t = threadIdx.x;
  for (int d = t; d < kD; d += 1024) rw[d] = rw_g[d];
  __syncthreads();
  for (int s = t; s < kS; s += 1024) {
    const float* xr = seq + ((long)b * kS + s) * kD;
    float a = 0.0f;
    for (int d = 0; d < kD; ++d) a += xr[d] * rw[d];
    vals[s] = a;
    idxs[s] = s;
  }
  __syncthreads();
  // block max over 2048
  float mx = fmaxf(vals[t], vals[t + 1024]);
#pragma unroll
  for (int m = 16; m >= 1; m >>= 1) mx = fmaxf(mx, __shfl_xor(mx, m));
  if ((t & 31) == 0) red[t >> 5] = mx;
  __syncthreads();
  float gm = -1e30f;
#pragma unroll
  for (int i = 0; i < 32; ++i) gm = fmaxf(gm, red[i]);
  __syncthreads();
  float e0 = __expf(vals[t] - gm), e1 = __expf(vals[t + 1024] - gm);
  float sum = e0 + e1;
#pragma unroll
  for (int m = 16; m >= 1; m >>= 1) sum += __shfl_xor(sum, m);
  if ((t & 31) == 0) red[t >> 5] = sum;
  __syncthreads();
  float Z = 0.0f;
#pragma unroll
  for (int i = 0; i < 32; ++i) Z += red[i];
  __syncthreads();
  vals[t] = e0;
  vals[t + 1024] = e1;
  __syncthreads();
  // bitonic sort, descending by (value, then smaller index)
  for (unsigned k = 2; k <= (unsigned)kS; k <<= 1) {
    for (unsigned j = k >> 1; j > 0; j >>= 1) {
      unsigned i = ((unsigned)t / j) * (2u * j) + ((unsigned)t % j);
      unsigned p = i + j;
      bool desc = ((i & k) == 0);
      float av = vals[i], bv = vals[p];
      int ai = idxs[i], bi = idxs[p];
      bool p_ranks_higher = (bv > av) || (bv == av && bi < ai);
      bool sw = desc ? p_ranks_higher : !p_ranks_higher;
      if (sw) { vals[i] = bv; vals[p] = av; idxs[i] = bi; idxs[p] = ai; }
      __syncthreads();
    }
  }
  if (t < kCAP) {
    topw[(long)b * kCAP + t] = vals[t] / Z;
    topi[(long)b * kCAP + t] = idxs[t];
  }
}

// ---------------------------------------------------------------------------
// LayerNorm (optional gather) -> bf16
// ---------------------------------------------------------------------------
DEV_INLINE float block_sum_256(float v, float* red) {
#pragma unroll
  for (int m = 16; m >= 1; m >>= 1) v += __shfl_xor(v, m);
  if ((threadIdx.x & 31) == 0) red[threadIdx.x >> 5] = v;
  __syncthreads();
  float tot = 0.0f;
#pragma unroll
  for (int i = 0; i < 8; ++i) tot += red[i];
  __syncthreads();
  return tot;
}

__global__ __launch_bounds__(256) void ln_gather_bf16_kernel(
    const float* __restrict__ x, const int* __restrict__ gidx,
    const float* __restrict__ gamma, const float* __restrict__ beta,
    u16* __restrict__ out, int rowsPerBatch) {
  __shared__ float red[8];
  const long orow = blockIdx.x;
  const long b = orow / rowsPerBatch;
  const long c = orow % rowsPerBatch;
  const long srow = gidx ? (b * kS + gidx[orow]) : (b * (long)kS + c);
  const float* xr = x + srow * kD;
  const int t = threadIdx.x;
  float lx[4];
  float s = 0.0f;
#pragma unroll
  for (int i = 0; i < 4; ++i) { lx[i] = xr[t + 256 * i]; s += lx[i]; }
  float mu = block_sum_256(s, red) * (1.0f / kD);
  float vs = 0.0f;
#pragma unroll
  for (int i = 0; i < 4; ++i) { float d = lx[i] - mu; vs += d * d; }
  float var = block_sum_256(vs, red) * (1.0f / kD);
  float inv = rsqrtf(var + kLnEps);
  bf16* o = (bf16*)out + orow * kD;
#pragma unroll
  for (int i = 0; i < 4; ++i) {
    int d = t + 256 * i;
    o[d] = (bf16)((lx[i] - mu) * inv * gamma[d] + beta[d]);
  }
}

// ---------------------------------------------------------------------------
// Tiled bf16 WMMA GEMM with double-buffered async global->LDS tile copies:
// C[M,N] = A[M,K] @ W[N,K]^T (+bias). All dims multiples of the tile sizes.
// ---------------------------------------------------------------------------
constexpr int GBM = 128, GBN = 128, GBK = 32, GLD = 40;

__global__ __launch_bounds__(256) void gemm_bf16_wmma_kernel(
    const u16* __restrict__ Aus, long aBatch, int lda,
    const u16* __restrict__ Wus, int ldw,
    float* __restrict__ C, long cBatch, int ldc,
    const float* __restrict__ bias, int K) {
  __shared__ bf16 sA[2][GBM][GLD];
  __shared__ bf16 sB[2][GBN][GLD];
  const bf16* A = (const bf16*)Aus + (long)blockIdx.z * aBatch;
  const bf16* W = (const bf16*)Wus;
  float* Cb = C + (long)blockIdx.z * cBatch;

  const int tid = threadIdx.x;
  const int lane = tid & 31;
  const int wave = tid >> 5;
  const int wm = wave >> 1;  // 0..3
  const int wn = wave & 1;   // 0..1
  const int mBlk = blockIdx.y * GBM;
  const int nBlk = blockIdx.x * GBN;
  const int lrow = tid >> 1;        // 0..127
  const int lcol = (tid & 1) * 16;  // 0 / 16
  const int rsel = lane & 15;
  const int koff = (lane < 16) ? 0 : 8;

  f32x8 acc[2][4];
#pragma unroll
  for (int i = 0; i < 2; ++i)
#pragma unroll
    for (int j = 0; j < 4; ++j) acc[i][j] = zero8();

  const bf16* gA = A + (long)(mBlk + lrow) * lda + lcol;
  const bf16* gW = W + (long)(nBlk + lrow) * ldw + lcol;

  // prologue: async-copy tile 0 into buffer 0 (4 x b128 per thread)
  async_ld_b128(lds32(&sA[0][lrow][lcol]), gA);
  async_ld_b128(lds32(&sA[0][lrow][lcol + 8]), gA + 8);
  async_ld_b128(lds32(&sB[0][lrow][lcol]), gW);
  async_ld_b128(lds32(&sB[0][lrow][lcol + 8]), gW + 8);

  const int nIter = K / GBK;
  for (int it = 0; it < nIter; ++it) {
    const int cur = it & 1;
    if (it + 1 < nIter) {
      const int nxt = cur ^ 1;
      const int k0 = (it + 1) * GBK;
      async_ld_b128(lds32(&sA[nxt][lrow][lcol]), gA + k0);
      async_ld_b128(lds32(&sA[nxt][lrow][lcol + 8]), gA + k0 + 8);
      async_ld_b128(lds32(&sB[nxt][lrow][lcol]), gW + k0);
      async_ld_b128(lds32(&sB[nxt][lrow][lcol + 8]), gW + k0 + 8);
      // own 4 "cur" copies retired once <=4 remain outstanding (in-order)
      asm volatile("s_wait_asynccnt 0x4" ::: "memory");
    } else {
      asm volatile("s_wait_asynccnt 0x0" ::: "memory");
    }
    __syncthreads();  // all waves' "cur" tile data resident in LDS

    bf16x16 afr[2], bfr[4];
#pragma unroll
    for (int tm = 0; tm < 2; ++tm)
      afr[tm] = make_frag(&sA[cur][wm * 32 + tm * 16 + rsel][0], koff);
#pragma unroll
    for (int tn = 0; tn < 4; ++tn)
      bfr[tn] = make_frag(&sB[cur][wn * 64 + tn * 16 + rsel][0], koff);
#pragma unroll
    for (int tm = 0; tm < 2; ++tm)
#pragma unroll
      for (int tn = 0; tn < 4; ++tn)
        acc[tm][tn] = WMMA_BF16(afr[tm], bfr[tn], acc[tm][tn]);
    __syncthreads();  // everyone done reading "cur" before it is overwritten
  }

#pragma unroll
  for (int tm = 0; tm < 2; ++tm)
#pragma unroll
    for (int tn = 0; tn < 4; ++tn)
#pragma unroll
      for (int r = 0; r < 8; ++r) {
        int row = mBlk + wm * 32 + tm * 16 + ((lane < 16) ? r : r + 8);
        int col = nBlk + wn * 64 + tn * 16 + rsel;
        float v = acc[tm][tn][r];
        if (bias) v += bias[col];
        Cb[(long)row * ldc + col] = v;
      }
}

// ---------------------------------------------------------------------------
// RoPE + bf16 conversion
// ---------------------------------------------------------------------------
__global__ void rope_q_bf16_kernel(const float* __restrict__ q,
                                   const int* __restrict__ topi,
                                   u16* __restrict__ qr) {
  const long total = (long)kB * kCAP * kD;
  const long stride = (long)gridDim.x * blockDim.x;
  for (long i = (long)blockIdx.x * blockDim.x + threadIdx.x; i < total; i += stride) {
    int d = (int)(i % kD);
    long row = i / kD;
    int hd = d & (kHD - 1);
    int j = hd & 31;
    float freq = __expf((float)j * (-kLog10000 / 31.0f));
    float ang = (float)topi[row] * freq;
    float rv = (hd < 32) ? __sinf(ang) : __cosf(ang);
    ((bf16*)qr)[i] = (bf16)(q[i] * rv);
  }
}

__global__ void rope_kv_bf16_kernel(const float* __restrict__ kv,
                                    u16* __restrict__ kr, u16* __restrict__ vb) {
  const long total = (long)kB * kS * 2 * kD;
  const long stride = (long)gridDim.x * blockDim.x;
  for (long i = (long)blockIdx.x * blockDim.x + threadIdx.x; i < total; i += stride) {
    int d2 = (int)(i % (2 * kD));
    long row = i / (2 * kD);  // b*S + s
    int s = (int)(row % kS);
    float v = kv[i];
    if (d2 < kD) {
      int hd = d2 & (kHD - 1);
      int j = hd & 31;
      float freq = __expf((float)j * (-kLog10000 / 31.0f));
      float ang = (float)s * freq;
      float rv = (hd < 32) ? __sinf(ang) : __cosf(ang);
      ((bf16*)kr)[row * kD + d2] = (bf16)(v * rv);
    } else {
      ((bf16*)vb)[row * kD + (d2 - kD)] = (bf16)v;
    }
  }
}

// ---------------------------------------------------------------------------
// Flash attention: per block 128 q-rows of one (b,h); S tiles of 64.
// Q and K tiles staged with async global->LDS copies; V transposed manually.
// ---------------------------------------------------------------------------
constexpr int FQ = 128, FS = 64, FP = 72;

__global__ __launch_bounds__(256) void flash_attn_kernel(
    const u16* __restrict__ qus, const u16* __restrict__ kus,
    const u16* __restrict__ vus, float* __restrict__ att) {
  __shared__ bf16 sQ[FQ][FP];
  __shared__ bf16 sK[FS][FP];
  __shared__ bf16 sVt[kHD][FP];
  __shared__ bf16 sP[8][16][FP];

  const int b = blockIdx.z, h = blockIdx.y;
  const int q0 = blockIdx.x * FQ;
  const bf16* Q = (const bf16*)qus + ((long)b * kCAP) * kD + h * kHD;
  const bf16* Kp = (const bf16*)kus + ((long)b * kS) * kD + h * kHD;
  const bf16* Vp = (const bf16*)vus + ((long)b * kS) * kD + h * kHD;
  const int tid = threadIdx.x, lane = tid & 31, wave = tid >> 5;
  const int rsel = lane & 15;
  const int koff = (lane < 16) ? 0 : 8;

  {
    int row = tid >> 1, cs = (tid & 1) * 32;
    const bf16* g = Q + (long)(q0 + row) * kD + cs;
#pragma unroll
    for (int i = 0; i < 32; i += 8)
      async_ld_b128(lds32(&sQ[row][cs + i]), g + i);
  }

  f32x8 oacc[4];
#pragma unroll
  for (int i = 0; i < 4; ++i) oacc[i] = zero8();
  float mrun[8], lrun[8];
#pragma unroll
  for (int r = 0; r < 8; ++r) { mrun[r] = -1e30f; lrun[r] = 0.0f; }

  const int krow = tid >> 2, kseg = (tid & 3) * 16;

  for (int s0 = 0; s0 < kS; s0 += FS) {
    {
      const bf16* gk = Kp + (long)(s0 + krow) * kD + kseg;
      async_ld_b128(lds32(&sK[krow][kseg]), gk);
      async_ld_b128(lds32(&sK[krow][kseg + 8]), gk + 8);
      const bf16* gv = Vp + (long)(s0 + krow) * kD + kseg;
      bf16 tmp[16];
      *(bf16x8*)tmp = *(const bf16x8*)gv;
      *(bf16x8*)(tmp + 8) = *(const bf16x8*)(gv + 8);
#pragma unroll
      for (int i = 0; i < 16; ++i) sVt[kseg + i][krow] = tmp[i];
    }
    asm volatile("s_wait_asynccnt 0x0" ::: "memory");
    __syncthreads();

    // scores: 16 q-rows (this wave) x 64 s-cols
    f32x8 sacc[4];
#pragma unroll
    for (int i = 0; i < 4; ++i) sacc[i] = zero8();
    const int qrow = wave * 16 + rsel;
#pragma unroll
    for (int kk = 0; kk < kHD; kk += 32) {
      bf16x16 af = make_frag(&sQ[qrow][kk], koff);
#pragma unroll
      for (int tn = 0; tn < 4; ++tn) {
        bf16x16 bfr = make_frag(&sK[tn * 16 + rsel][kk], koff);
        sacc[tn] = WMMA_BF16(af, bfr, sacc[tn]);
      }
    }
#pragma unroll
    for (int tn = 0; tn < 4; ++tn)
#pragma unroll
      for (int r = 0; r < 8; ++r) sacc[tn][r] *= 0.125f;  // 1/sqrt(64)

    const int prow = (lane < 16) ? 0 : 8;
#pragma unroll
    for (int r = 0; r < 8; ++r) {
      float mx = fmaxf(fmaxf(sacc[0][r], sacc[1][r]), fmaxf(sacc[2][r], sacc[3][r]));
#pragma unroll
      for (int m = 8; m >= 1; m >>= 1) mx = fmaxf(mx, __shfl_xor(mx, m));
      float mnew = fmaxf(mrun[r], mx);
      float esc = __expf(mrun[r] - mnew);
      float psum = 0.0f;
#pragma unroll
      for (int tn = 0; tn < 4; ++tn) {
        float p = __expf(sacc[tn][r] - mnew);
        sP[wave][prow + r][tn * 16 + rsel] = (bf16)p;
        psum += p;
      }
#pragma unroll
      for (int m = 8; m >= 1; m >>= 1) psum += __shfl_xor(psum, m);
      lrun[r] = lrun[r] * esc + psum;
      mrun[r] = mnew;
#pragma unroll
      for (int tn = 0; tn < 4; ++tn) oacc[tn][r] *= esc;
    }
    // same-wave LDS RAW on sP: wait for ds stores before fragment loads
    asm volatile("s_wait_dscnt 0x0" ::: "memory");

#pragma unroll
    for (int kk = 0; kk < FS; kk += 32) {
      bf16x16 af = make_frag(&sP[wave][rsel][kk], koff);
#pragma unroll
      for (int tn = 0; tn < 4; ++tn) {
        bf16x16 bfr = make_frag(&sVt[tn * 16 + rsel][kk], koff);
        oacc[tn] = WMMA_BF16(af, bfr, oacc[tn]);
      }
    }
    __syncthreads();
  }

  float* O = att + ((long)b * kCAP + q0) * kD + h * kHD;
#pragma unroll
  for (int tn = 0; tn < 4; ++tn)
#pragma unroll
    for (int r = 0; r < 8; ++r) {
      int rowl = wave * 16 + ((lane < 16) ? r : r + 8);
      int col = tn * 16 + rsel;
      O[(long)rowl * kD + col] = oacc[tn][r] / lrun[r];
    }
}

// ---------------------------------------------------------------------------
// Elementwise helpers
// ---------------------------------------------------------------------------
__global__ void f32_to_bf16_kernel(const float* __restrict__ x, u16* __restrict__ o, long n) {
  const long stride = (long)gridDim.x * blockDim.x;
  for (long i = (long)blockIdx.x * blockDim.x + threadIdx.x; i < n; i += stride)
    ((bf16*)o)[i] = (bf16)x[i];
}

__global__ void swiglu_bf16_kernel(const float* __restrict__ h, u16* __restrict__ g) {
  const long total = (long)kB * kCAP * kD;
  const long stride = (long)gridDim.x * blockDim.x;
  for (long i = (long)blockIdx.x * blockDim.x + threadIdx.x; i < total; i += stride) {
    long row = i / kD;
    int d = (int)(i % kD);
    float x1 = h[row * 2 * kD + d];
    float gt = h[row * 2 * kD + kD + d];
    float si = gt / (1.0f + __expf(-gt));
    ((bf16*)g)[i] = (bf16)(si * x1);
  }
}

__global__ void copy_f32_kernel(const float* __restrict__ x, float* __restrict__ o, long n) {
  const long stride = (long)gridDim.x * blockDim.x;
  for (long i = (long)blockIdx.x * blockDim.x + threadIdx.x; i < n; i += stride) o[i] = x[i];
}

__global__ void combine_scatter_kernel(const float* __restrict__ atto,
                                       const float* __restrict__ ffn,
                                       const float* __restrict__ topw,
                                       const int* __restrict__ topi,
                                       float* __restrict__ out) {
  const long total = (long)kB * kCAP * kD;
  const long stride = (long)gridDim.x * blockDim.x;
  for (long i = (long)blockIdx.x * blockDim.x + threadIdx.x; i < total; i += stride) {
    long d = i % kD;
    long row = i / kD;  // b*CAP + c
    long b = row / kCAP;
    int s = topi[row];
    out[((long)b * kS + s) * kD + d] += (atto[i] + ffn[i]) * topw[row];
  }
}

// ---------------------------------------------------------------------------
// Host driver
// ---------------------------------------------------------------------------
extern "C" void kernel_launch(void* const* d_in, const int* in_sizes, int n_in,
                              void* d_out, int out_size, void* d_ws, size_t ws_size,
                              hipStream_t stream) {
  (void)in_sizes; (void)n_in; (void)out_size; (void)ws_size;
  const float* seq     = (const float*)d_in[0];
  const float* router_w= (const float*)d_in[1];
  const float* qln_g   = (const float*)d_in[2];
  const float* qln_b   = (const float*)d_in[3];
  const float* vln_g   = (const float*)d_in[4];
  const float* vln_b   = (const float*)d_in[5];
  const float* q_w     = (const float*)d_in[6];
  const float* kv_w    = (const float*)d_in[7];
  const float* out_w   = (const float*)d_in[8];
  const float* fc1_w   = (const float*)d_in[9];
  const float* fc1_b   = (const float*)d_in[10];
  const float* fc2_w   = (const float*)d_in[11];
  const float* fc2_b   = (const float*)d_in[12];
  float* out = (float*)d_out;

  char* ws = (char*)d_ws;
  size_t off = 0;
  auto alloc = [&](size_t bytes) -> char* {
    char* p = ws + off;
    off += (bytes + 255) & ~(size_t)255;
    return p;
  };
  float* topw = (float*)alloc((size_t)kB * kCAP * 4);
  int*   topi = (int*)  alloc((size_t)kB * kCAP * 4);
  u16* wq   = (u16*)alloc((size_t)kD * kD * 2);
  u16* wkv  = (u16*)alloc((size_t)2 * kD * kD * 2);
  u16* wout = (u16*)alloc((size_t)kD * kD * 2);
  u16* wfc1 = (u16*)alloc((size_t)2 * kD * kD * 2);
  u16* wfc2 = (u16*)alloc((size_t)kD * kD * 2);
  u16* qs = (u16*)alloc((size_t)kB * kCAP * kD * 2);
  u16* vs = (u16*)alloc((size_t)kB * kS * kD * 2);
  u16* qr = (u16*)alloc((size_t)kB * kCAP * kD * 2);
  u16* kr = (u16*)alloc((size_t)kB * kS * kD * 2);
  u16* vb = (u16*)alloc((size_t)kB * kS * kD * 2);
  // Region A: q_f32, later att_f32, later ffn_f32 (each B*CAP*D f32)
  float* regA = (float*)alloc((size_t)kB * kCAP * kD * 4);
  // Region B: kv_f32 (B*S*2D f32); reused after RoPE for att_bf/att_o/h/gate
  char* regB = alloc((size_t)kB * kS * 2 * kD * 4);
  float* qf = regA;
  float* attf = regA;
  float* ffnf = regA;
  float* kvf = (float*)regB;
  u16*   attbf = (u16*)regB;                                        // +0   (8MB)
  float* atto = (float*)(regB + (size_t)kB * kCAP * kD * 2);        // +8MB (16MB)
  float* hbuf = (float*)(regB + (size_t)kB * kCAP * kD * 2
                              + (size_t)kB * kCAP * kD * 4);        // +24MB (32MB)
  u16*   gbf  = (u16*)(regB + (size_t)kB * kCAP * kD * 2
                            + (size_t)kB * kCAP * kD * 4
                            + (size_t)kB * kCAP * 2 * kD * 4);      // +56MB (8MB)

  // 1. router + top-k
  router_topk_kernel<<<kB, 1024, 0, stream>>>(seq, router_w, topw, topi);

  // 2. weights -> bf16
  auto cvt = [&](const float* src, u16* dst, long n) {
    unsigned blocks = (unsigned)((n + 1023) / 1024);
    f32_to_bf16_kernel<<<blocks, 256, 0, stream>>>(src, dst, n);
  };
  cvt(q_w, wq, (long)kD * kD);
  cvt(kv_w, wkv, (long)2 * kD * kD);
  cvt(out_w, wout, (long)kD * kD);
  cvt(fc1_w, wfc1, (long)2 * kD * kD);
  cvt(fc2_w, wfc2, (long)kD * kD);

  // 3. gather + LN (queries), LN (values)
  ln_gather_bf16_kernel<<<kB * kCAP, 256, 0, stream>>>(seq, topi, qln_g, qln_b, qs, kCAP);
  ln_gather_bf16_kernel<<<kB * kS, 256, 0, stream>>>(seq, nullptr, vln_g, vln_b, vs, kS);

  // 4. q / kv projections (WMMA GEMM, async double-buffered tiles)
  gemm_bf16_wmma_kernel<<<dim3(kD / GBN, kCAP / GBM, kB), 256, 0, stream>>>(
      qs, (long)kCAP * kD, kD, wq, kD, qf, (long)kCAP * kD, kD, nullptr, kD);
  gemm_bf16_wmma_kernel<<<dim3(2 * kD / GBN, kS / GBM, kB), 256, 0, stream>>>(
      vs, (long)kS * kD, kD, wkv, kD, kvf, (long)kS * 2 * kD, 2 * kD, nullptr, kD);

  // 5. RoPE + bf16
  rope_q_bf16_kernel<<<4096, 256, 0, stream>>>(qf, topi, qr);
  rope_kv_bf16_kernel<<<16384, 256, 0, stream>>>(kvf, kr, vb);

  // 6. flash attention (WMMA) -> att f32 (reuses region A)
  flash_attn_kernel<<<dim3(kCAP / FQ, kH, kB), 256, 0, stream>>>(qr, kr, vb, attf);

  // 7. out projection
  f32_to_bf16_kernel<<<16384, 256, 0, stream>>>(attf, attbf, (long)kB * kCAP * kD);
  gemm_bf16_wmma_kernel<<<dim3(kD / GBN, kCAP / GBM, kB), 256, 0, stream>>>(
      attbf, (long)kCAP * kD, kD, wout, kD, atto, (long)kCAP * kD, kD, nullptr, kD);

  // 8. FFN: fc1 (+bias), SwiGLU, fc2 (+bias) -> ffn f32 (reuses region A)
  gemm_bf16_wmma_kernel<<<dim3(2 * kD / GBN, kCAP / GBM, kB), 256, 0, stream>>>(
      qs, (long)kCAP * kD, kD, wfc1, kD, hbuf, (long)kCAP * 2 * kD, 2 * kD, fc1_b, kD);
  swiglu_bf16_kernel<<<4096, 256, 0, stream>>>(hbuf, gbf);
  gemm_bf16_wmma_kernel<<<dim3(kD / GBN, kCAP / GBM, kB), 256, 0, stream>>>(
      gbf, (long)kCAP * kD, kD, wfc2, kD, ffnf, (long)kCAP * kD, kD, fc2_b, kD);

  // 9. combine: out = seq; out[b, top_idx] += (att + ffn) * top_w
  copy_f32_kernel<<<8192, 256, 0, stream>>>(seq, out, (long)kB * kS * kD);
  combine_scatter_kernel<<<4096, 256, 0, stream>>>(atto, ffnf, topw, topi, out);
}
